// FeatureAlignmentModule_73624329388232
// MI455X (gfx1250) — matrix operand order repaired and checked by reading, not compiled, not run on hardware
//
#include <hip/hip_runtime.h>
#include <hip/hip_bf16.h>

typedef __attribute__((ext_vector_type(16))) _Float16 v16h;
typedef __attribute__((ext_vector_type(8)))  float    v8f;
typedef __attribute__((ext_vector_type(4)))  int      v4i;

#define DIM  1024
#define NSRC 4096
#define NTGT 4096

// ---------------------------------------------------------------------------
// CDNA5 async global->LDS copy (ASYNCcnt-tracked), with portable fallback
// Builtin signature (from hipcc diagnostic): param0 = v4i addrspace(1)*,
// param1 = v4i addrspace(3)*, then imm offset, imm cpol.
// ---------------------------------------------------------------------------
#if __has_builtin(__builtin_amdgcn_global_load_async_to_lds_b128)
#define ASYNC_LDS 1
#else
#define ASYNC_LDS 0
#endif

typedef __attribute__((address_space(1))) v4i gas_v4i;
typedef __attribute__((address_space(3))) v4i las_v4i;

__device__ __forceinline__ void async_copy_b128(void* lds_dst, const void* gsrc) {
#if ASYNC_LDS
  __builtin_amdgcn_global_load_async_to_lds_b128(
      (gas_v4i*)gsrc, (las_v4i*)lds_dst, 0, 0);
#else
  *(uint4*)lds_dst = *(const uint4*)gsrc;
#endif
}

__device__ __forceinline__ void wait_async_copies() {
#if ASYNC_LDS
#if __has_builtin(__builtin_amdgcn_s_wait_asynccnt)
  __builtin_amdgcn_s_wait_asynccnt(0);
#else
  asm volatile("s_wait_asynccnt 0" ::: "memory");
#endif
#endif
}

// Pack two f32 into a dword of two f16
__device__ __forceinline__ unsigned pack2h(float a, float b) {
  union { _Float16 h[2]; unsigned u; } p;
  p.h[0] = (_Float16)a;
  p.h[1] = (_Float16)b;
  return p.u;
}

// ---------------------------------------------------------------------------
// Fragment loaders following CDNA5 ISA WMMA VGPR layouts (CDNA5 ISA §7.12.2)
// ---------------------------------------------------------------------------

// A-matrix 16x32 f16, LDS tile row-major [m][k] stride lda:
// lanes 0-15 rows M=0..15, lanes 16-31 same rows, K-groups +8
__device__ __forceinline__ v16h load_frag_a(const _Float16* As, int lda,
                                            int rowBase, int lane) {
  const _Float16* p = As + (rowBase + (lane & 15)) * lda;
  const int g8 = (lane >> 4) * 8;
  v16h a;
#pragma unroll
  for (int v = 0; v < 4; ++v) {
    int k0 = 2 * v + g8;
    a[2 * v]     = p[k0];
    a[2 * v + 1] = p[k0 + 1];
    int k1 = 16 + 2 * v + g8;
    a[8 + 2 * v]     = p[k1];
    a[8 + 2 * v + 1] = p[k1 + 1];
  }
  return a;
}

// B-matrix 32x16 f16, LDS tile stored [k][n] with ldn=128:
// lane%16 = N, elements = K in table order (lanes 16-31 K-groups +8)
__device__ __forceinline__ v16h load_frag_b_kn(const _Float16* Bs, int nBase,
                                               int lane) {
  const _Float16* p = Bs + nBase + (lane & 15);
  const int g8 = (lane >> 4) * 8;
  v16h b;
#pragma unroll
  for (int h = 0; h < 16; ++h) b[h] = p[((h < 8) ? (g8 + h) : (16 + g8 + h - 8)) * 128];
  return b;
}

__device__ __forceinline__ v8f wmma_f16(v16h a, v16h b, v8f c) {
  return __builtin_amdgcn_wmma_f32_16x16x32_f16(false, a, false, b,
                                                (short)0, c, false, false);
}

// ---------------------------------------------------------------------------
// f32 -> f16 conversion (packed dword stores)
// ---------------------------------------------------------------------------
__global__ void __launch_bounds__(256) cvt_f32_to_f16(const float* __restrict__ in,
                                                      _Float16* __restrict__ out,
                                                      int n) {
  int i = (blockIdx.x * blockDim.x + threadIdx.x) * 4;
  if (i < n) {
    float4 v = *(const float4*)(in + i);
    uint2 o;
    o.x = pack2h(v.x, v.y);
    o.y = pack2h(v.z, v.w);
    *(uint2*)(out + i) = o;
  }
}

// ---------------------------------------------------------------------------
// Projection: P = X @ W^T + bias   (X:[M][K] f16, W:[N][K] f16, P:[M][N] f16)
// Block tile 128x128, 8 waves each computing 64x32 (4x2 wmma accumulators)
// A tile: async copy (layout preserved). B tile: transposed into [k][n].
// ---------------------------------------------------------------------------
__global__ void __launch_bounds__(256) proj_gemm_kernel(
    const _Float16* __restrict__ X, const _Float16* __restrict__ W,
    const float* __restrict__ bias, _Float16* __restrict__ P,
    int M, int N, int K) {
  __shared__ __align__(16) _Float16 As[128 * 32];   // [m][k]
  __shared__ __align__(16) _Float16 Bs[32 * 128];   // [k][n]
  const int t = threadIdx.x;
  const int lane = t & 31;
  const int wave = t >> 5;
  const int wm = wave >> 2;  // 0..1
  const int wn = wave & 3;   // 0..3
  const int m0 = blockIdx.y * 128;
  const int n0 = blockIdx.x * 128;

  const v8f vzero = {0.f, 0.f, 0.f, 0.f, 0.f, 0.f, 0.f, 0.f};
  v8f acc[4][2];
#pragma unroll
  for (int i = 0; i < 4; ++i)
#pragma unroll
    for (int j = 0; j < 2; ++j) acc[i][j] = vzero;

  const int r = t >> 1, seg = (t & 1) * 16;
  for (int k0 = 0; k0 < K; k0 += 32) {
    // A tile 128x32: pure copy, async to LDS
    async_copy_b128(As + r * 32 + seg,     X + (size_t)(m0 + r) * K + k0 + seg);
    async_copy_b128(As + r * 32 + seg + 8, X + (size_t)(m0 + r) * K + k0 + seg + 8);
    // B tile: row n of W (contiguous in k) transposed into Bs[k][n]
    {
      const uint4* srcp = (const uint4*)(W + (size_t)(n0 + r) * K + k0 + seg);
      uint4 d0 = srcp[0], d1 = srcp[1];
      const _Float16* h0 = (const _Float16*)&d0;
      const _Float16* h1 = (const _Float16*)&d1;
#pragma unroll
      for (int q = 0; q < 8; ++q) Bs[(seg + q) * 128 + r] = h0[q];
#pragma unroll
      for (int q = 0; q < 8; ++q) Bs[(seg + 8 + q) * 128 + r] = h1[q];
    }
    wait_async_copies();
    __syncthreads();

    v16h af[4], bf[2];
#pragma unroll
    for (int i = 0; i < 4; ++i) af[i] = load_frag_a(As, 32, wm * 64 + i * 16, lane);
#pragma unroll
    for (int j = 0; j < 2; ++j) bf[j] = load_frag_b_kn(Bs, wn * 32 + j * 16, lane);
#pragma unroll
    for (int i = 0; i < 4; ++i)
#pragma unroll
      for (int j = 0; j < 2; ++j) acc[i][j] = wmma_f16(af[i], bf[j], acc[i][j]);
    __syncthreads();
  }

  const int colLane = lane & 15;
  const int rowOff = (lane >> 4) * 8;
#pragma unroll
  for (int i = 0; i < 4; ++i)
#pragma unroll
    for (int j = 0; j < 2; ++j) {
      int col = n0 + wn * 32 + j * 16 + colLane;
      float bv = bias[col];
#pragma unroll
      for (int v = 0; v < 8; ++v) {
        int row = m0 + wm * 64 + i * 16 + rowOff + v;
        P[(size_t)row * N + col] = (_Float16)(acc[i][j][v] + bv);
      }
    }
}

// ---------------------------------------------------------------------------
// Per-row squared norm of an f16 matrix (deterministic wave reduction)
// ---------------------------------------------------------------------------
__global__ void __launch_bounds__(256) rowsq_kernel(const _Float16* __restrict__ P,
                                                    float* __restrict__ sq, int K) {
  int row = blockIdx.x * 8 + (threadIdx.x >> 5);
  int lane = threadIdx.x & 31;
  const _Float16* p = P + (size_t)row * K;
  float s = 0.f;
  for (int c = lane; c < K; c += 32) {
    float x = (float)p[c];
    s += x * x;
  }
#pragma unroll
  for (int off = 16; off > 0; off >>= 1) s += __shfl_down(s, off, 32);
  if (lane == 0) sq[row] = s;
}

// ---------------------------------------------------------------------------
// E = exp(-cdist(sp,tp)/T): GEMM sp @ tp^T fused with distance epilogue
// ---------------------------------------------------------------------------
__global__ void __launch_bounds__(256) dist_kernel(
    const _Float16* __restrict__ SP, const _Float16* __restrict__ TP,
    const float* __restrict__ sqs, const float* __restrict__ sqt,
    const float* __restrict__ tempPtr, float* __restrict__ E) {
  __shared__ __align__(16) _Float16 As[128 * 32];   // [m][k]
  __shared__ __align__(16) _Float16 Bs[32 * 128];   // [k][n]
  const int t = threadIdx.x;
  const int lane = t & 31;
  const int wave = t >> 5;
  const int wm = wave >> 2;
  const int wn = wave & 3;
  const int m0 = blockIdx.y * 128;
  const int n0 = blockIdx.x * 128;

  const v8f vzero = {0.f, 0.f, 0.f, 0.f, 0.f, 0.f, 0.f, 0.f};
  v8f acc[4][2];
#pragma unroll
  for (int i = 0; i < 4; ++i)
#pragma unroll
    for (int j = 0; j < 2; ++j) acc[i][j] = vzero;

  const int r = t >> 1, seg = (t & 1) * 16;
  for (int k0 = 0; k0 < DIM; k0 += 32) {
    async_copy_b128(As + r * 32 + seg,     SP + (size_t)(m0 + r) * DIM + k0 + seg);
    async_copy_b128(As + r * 32 + seg + 8, SP + (size_t)(m0 + r) * DIM + k0 + seg + 8);
    {
      const uint4* srcp = (const uint4*)(TP + (size_t)(n0 + r) * DIM + k0 + seg);
      uint4 d0 = srcp[0], d1 = srcp[1];
      const _Float16* h0 = (const _Float16*)&d0;
      const _Float16* h1 = (const _Float16*)&d1;
#pragma unroll
      for (int q = 0; q < 8; ++q) Bs[(seg + q) * 128 + r] = h0[q];
#pragma unroll
      for (int q = 0; q < 8; ++q) Bs[(seg + 8 + q) * 128 + r] = h1[q];
    }
    wait_async_copies();
    __syncthreads();

    v16h af[4], bf[2];
#pragma unroll
    for (int i = 0; i < 4; ++i) af[i] = load_frag_a(As, 32, wm * 64 + i * 16, lane);
#pragma unroll
    for (int j = 0; j < 2; ++j) bf[j] = load_frag_b_kn(Bs, wn * 32 + j * 16, lane);
#pragma unroll
    for (int i = 0; i < 4; ++i)
#pragma unroll
      for (int j = 0; j < 2; ++j) acc[i][j] = wmma_f16(af[i], bf[j], acc[i][j]);
    __syncthreads();
  }

  const float invT = -1.0f / tempPtr[0];
  const int colLane = lane & 15;
  const int rowOff = (lane >> 4) * 8;
#pragma unroll
  for (int i = 0; i < 4; ++i)
#pragma unroll
    for (int j = 0; j < 2; ++j) {
      int col = n0 + wn * 32 + j * 16 + colLane;
      float st = sqt[col];
#pragma unroll
      for (int v = 0; v < 8; ++v) {
        int row = m0 + wm * 64 + i * 16 + rowOff + v;
        float d2 = sqs[row] + st - 2.0f * acc[i][j][v];
        float cost = sqrtf(fmaxf(d2, 0.0f));
        E[(size_t)row * NTGT + col] = __expf(cost * invT);
      }
    }
}

// ---------------------------------------------------------------------------
// In-place row normalization (one wave per row, deterministic order)
// ---------------------------------------------------------------------------
__global__ void __launch_bounds__(256) rownorm_kernel(float* __restrict__ E, int N) {
  int row = blockIdx.x * 8 + (threadIdx.x >> 5);
  int lane = threadIdx.x & 31;
  float* p = E + (size_t)row * N;
  float s = 0.f;
  for (int c = lane; c < N; c += 32) s += p[c];
#pragma unroll
  for (int off = 16; off > 0; off >>= 1) s += __shfl_down(s, off, 32);
  s = __shfl(s, 0, 32);
  float inv = 1.0f / s;
  for (int c = lane; c < N; c += 32) p[c] *= inv;
}

// ---------------------------------------------------------------------------
// aligned = K @ target  (K:[4096][4096] f32, converted to f16 in staging;
// target pre-converted to f16, its tile copied async into [k][n] LDS)
// ---------------------------------------------------------------------------
__global__ void __launch_bounds__(256) aligned_kernel(
    const float* __restrict__ Kmat, const _Float16* __restrict__ T,
    float* __restrict__ Out) {
  __shared__ __align__(16) _Float16 As[128 * 32];   // [m][k]
  __shared__ __align__(16) _Float16 Bs[32 * 128];   // [k][n]
  const int t = threadIdx.x;
  const int lane = t & 31;
  const int wave = t >> 5;
  const int wm = wave >> 2;
  const int wn = wave & 3;
  const int m0 = blockIdx.y * 128;
  const int n0 = blockIdx.x * 128;

  const v8f vzero = {0.f, 0.f, 0.f, 0.f, 0.f, 0.f, 0.f, 0.f};
  v8f acc[4][2];
#pragma unroll
  for (int i = 0; i < 4; ++i)
#pragma unroll
    for (int j = 0; j < 2; ++j) acc[i][j] = vzero;

  const int r = t >> 1, seg = (t & 1) * 16;      // A staging: 16 floats/thread
  const int kb = t >> 3, nseg = (t & 7) * 16;    // B staging: 16 halves/thread
  for (int k0 = 0; k0 < NTGT; k0 += 32) {
    {  // A tile 128x32: f32 -> f16, packed dword stores to LDS
      const float4* srcp = (const float4*)(Kmat + (size_t)(m0 + r) * NTGT + k0 + seg);
      unsigned* dstu = (unsigned*)(As + r * 32 + seg);
#pragma unroll
      for (int q = 0; q < 4; ++q) {
        float4 v4 = srcp[q];
        dstu[q * 2 + 0] = pack2h(v4.x, v4.y);
        dstu[q * 2 + 1] = pack2h(v4.z, v4.w);
      }
    }
    // B tile 32x128: pure copy (target rows contiguous along n), async to LDS
    async_copy_b128(Bs + kb * 128 + nseg,     T + (size_t)(k0 + kb) * DIM + n0 + nseg);
    async_copy_b128(Bs + kb * 128 + nseg + 8, T + (size_t)(k0 + kb) * DIM + n0 + nseg + 8);
    wait_async_copies();
    __syncthreads();

    v16h af[4], bf[2];
#pragma unroll
    for (int i = 0; i < 4; ++i) af[i] = load_frag_a(As, 32, wm * 64 + i * 16, lane);
#pragma unroll
    for (int j = 0; j < 2; ++j) bf[j] = load_frag_b_kn(Bs, wn * 32 + j * 16, lane);
#pragma unroll
    for (int i = 0; i < 4; ++i)
#pragma unroll
      for (int j = 0; j < 2; ++j) acc[i][j] = wmma_f16(af[i], bf[j], acc[i][j]);
    __syncthreads();
  }

  const int colLane = lane & 15;
  const int rowOff = (lane >> 4) * 8;
#pragma unroll
  for (int i = 0; i < 4; ++i)
#pragma unroll
    for (int j = 0; j < 2; ++j) {
      int col = n0 + wn * 32 + j * 16 + colLane;
#pragma unroll
      for (int v = 0; v < 8; ++v) {
        int row = m0 + wm * 64 + i * 16 + rowOff + v;
        Out[(size_t)row * DIM + col] = acc[i][j][v];
      }
    }
}

// ---------------------------------------------------------------------------
extern "C" void kernel_launch(void* const* d_in, const int* in_sizes, int n_in,
                              void* d_out, int out_size, void* d_ws, size_t ws_size,
                              hipStream_t stream) {
  const float* source = (const float*)d_in[0];
  const float* target = (const float*)d_in[1];
  const float* W_src  = (const float*)d_in[2];
  const float* b_src  = (const float*)d_in[3];
  const float* W_tgt  = (const float*)d_in[4];
  const float* b_tgt  = (const float*)d_in[5];
  const float* temp   = (const float*)d_in[6];

  float* alignedOut = (float*)d_out;                       // [4096][1024]
  float* Kout = alignedOut + (size_t)NSRC * DIM;           // [4096][4096]

  _Float16* srcH = (_Float16*)d_ws;
  _Float16* tgtH = srcH + (size_t)NSRC * DIM;
  _Float16* WsH  = tgtH + (size_t)NTGT * DIM;
  _Float16* WtH  = WsH + (size_t)DIM * DIM;
  _Float16* spH  = WtH + (size_t)DIM * DIM;
  _Float16* tpH  = spH + (size_t)NSRC * DIM;
  float* sqs = (float*)(tpH + (size_t)NTGT * DIM);
  float* sqt = sqs + NSRC;

  // 1) f32 -> f16 operand copies
  cvt_f32_to_f16<<<(NSRC * DIM / 4) / 256, 256, 0, stream>>>(source, srcH, NSRC * DIM);
  cvt_f32_to_f16<<<(NTGT * DIM / 4) / 256, 256, 0, stream>>>(target, tgtH, NTGT * DIM);
  cvt_f32_to_f16<<<(DIM * DIM / 4) / 256, 256, 0, stream>>>(W_src, WsH, DIM * DIM);
  cvt_f32_to_f16<<<(DIM * DIM / 4) / 256, 256, 0, stream>>>(W_tgt, WtH, DIM * DIM);

  // 2) projections (WMMA f16 -> f32 accum), bias fused
  proj_gemm_kernel<<<dim3(DIM / 128, NSRC / 128), 256, 0, stream>>>(
      srcH, WsH, b_src, spH, NSRC, DIM, DIM);
  proj_gemm_kernel<<<dim3(DIM / 128, NTGT / 128), 256, 0, stream>>>(
      tgtH, WtH, b_tgt, tpH, NTGT, DIM, DIM);

  // 3) squared norms
  rowsq_kernel<<<NSRC / 8, 256, 0, stream>>>(spH, sqs, DIM);
  rowsq_kernel<<<NTGT / 8, 256, 0, stream>>>(tpH, sqt, DIM);

  // 4) distance GEMM fused with exp(-cost/T) -> unnormalized K
  dist_kernel<<<dim3(NTGT / 128, NSRC / 128), 256, 0, stream>>>(
      spH, tpH, sqs, sqt, temp, Kout);

  // 5) row-normalize K in place (deterministic, no atomics)
  rownorm_kernel<<<NSRC / 8, 256, 0, stream>>>(Kout, NTGT);

  // 6) aligned = K @ target
  aligned_kernel<<<dim3(DIM / 128, NSRC / 128), 256, 0, stream>>>(
      Kout, tgtH, alignedOut);
}
